// Qwen3VLWithExpertModel_31533649887533
// MI455X (gfx1250) — matrix operand order repaired and checked by reading, not compiled, not run on hardware
//
#include <hip/hip_runtime.h>
#include <hip/hip_bf16.h>
#include <math.h>

// ---------------- model constants ----------------
#define BB 2
#define HH 16
#define KHH 8
#define DHH 128
#define DD 2048
#define SS 1344
#define SEQ0 1024
#define SEQ1 256
#define SEQ2 64
#define EPSV 1e-6f
#define SCALEV 0.08838834764831845f   // 128^-0.5
#define LOGTHETA 13.815510557964274f  // ln(1e6)

// ---------------- WMMA types ----------------
typedef __attribute__((ext_vector_type(16))) __bf16 v16bf;
typedef __attribute__((ext_vector_type(8)))  float  v8f;

union BF16Frag { v16bf v; unsigned int u[8]; };

// pack two fp32 -> two bf16 in one dword (lowers to v_cvt_pk_bf16_f32)
__device__ __forceinline__ unsigned int packbf(float a, float b) {
  union { __bf16 h[2]; unsigned int u; } r;
  r.h[0] = (__bf16)a; r.h[1] = (__bf16)b;
  return r.u;
}

// CDNA5 async global->LDS copy, 16B per lane (ASYNCcnt-tracked)
__device__ __forceinline__ void async_ld128(unsigned lds_byte_off, const float* g) {
  asm volatile("global_load_async_to_lds_b128 %0, %1, off"
               :: "v"(lds_byte_off), "v"(g) : "memory");
}
__device__ __forceinline__ void wait_async0() {
#if __has_builtin(__builtin_amdgcn_s_wait_asynccnt)
  __builtin_amdgcn_s_wait_asynccnt(0);
#else
  asm volatile("s_wait_asynccnt 0x0" ::: "memory");
#endif
}

// ---------------- tiled bf16-WMMA GEMM ----------------
// C[M,N] = epi( A[M,K] @ B^T ).  BLAYOUT==0: B is [N,K] (K contiguous).
// BLAYOUT==1: B is [K,N] (N contiguous).  A,B fp32 in memory; tiles are
// DMA'd raw into double-buffered LDS via async global->LDS, converted to
// bf16 during fragment gather (ds_load_b128 + v_cvt_pk_bf16_f32),
// accumulated fp32 via v_wmma_f32_16x16x32_bf16.
// All dims assumed M%64==0, N%64==0, K%32==0 (true for every GEMM here).
// Batched over grid.z; B offset supports GQA:
//   offB = (z/hPerB)*strideBb + ((z%hPerB)/gqaDiv)*strideBh
// EPI: 0 none, 1 C=acc+E, 2 C=acc*scale+E (E has its own ld/stride/div).
#define TM 64
#define TN 64
#define TK 32
#define LDA_T (TK + 4)            // 36 floats per A row (144B, 16B-multiple)
#define LDB0  (TK + 4)            // layout0 B row stride
#define LDB1  (TN + 4)            // layout1 B row stride (68 floats = 272B)
#define BTILE (TN * (TK + 4))     // 2304 floats >= 32*68

template <int EPI, int BLAYOUT>
__global__ __launch_bounds__(128)
void gemm_kernel(const float* __restrict__ A, const float* __restrict__ Bw,
                 float* __restrict__ C, const float* __restrict__ E,
                 int K, int ldA, int ldB, int ldC, int ldE,
                 long long strideA, long long strideBb, long long strideBh,
                 int hPerB, int gqaDiv,
                 long long strideC, long long strideE, int eDiv,
                 float scale)
{
  __shared__ __align__(16) float As[2][TM * LDA_T];
  __shared__ __align__(16) float Bs[2][BTILE];

  const int tid = threadIdx.x;
  const int z = blockIdx.z;
  const int m0 = blockIdx.y * TM;
  const int n0 = blockIdx.x * TN;

  A  += (size_t)z * strideA;
  Bw += (size_t)(z / hPerB) * strideBb + (size_t)((z % hPerB) / gqaDiv) * strideBh;
  C  += (size_t)z * strideC;
  if (EPI) E += (size_t)(z / eDiv) * strideE;

  const int wave = tid >> 5;
  const int lane = tid & 31;
  const int half = lane >> 4;
  const int lr   = lane & 15;
  const int wm = (wave >> 1) * 32;
  const int wn = (wave & 1) * 32;

  v8f acc[2][2] = {};

  auto issueTile = [&](int kt, int buf) {
    // A tile: 64 rows x 32 k, 512 x 16B chunks over 128 threads
#pragma unroll
    for (int it = 0; it < 4; ++it) {
      int idx = tid + it * 128;
      int row = idx >> 3;
      int kc  = (idx & 7) << 2;
      async_ld128((unsigned)(size_t)&As[buf][row * LDA_T + kc],
                  A + (size_t)(m0 + row) * ldA + (size_t)kt * TK + kc);
    }
    if (BLAYOUT == 0) {
#pragma unroll
      for (int it = 0; it < 4; ++it) {
        int idx = tid + it * 128;
        int row = idx >> 3;
        int kc  = (idx & 7) << 2;
        async_ld128((unsigned)(size_t)&Bs[buf][row * LDB0 + kc],
                    Bw + (size_t)(n0 + row) * ldB + (size_t)kt * TK + kc);
      }
    } else {
#pragma unroll
      for (int it = 0; it < 4; ++it) {
        int idx = tid + it * 128;
        int k  = idx >> 4;                 // 0..31
        int nc = (idx & 15) << 2;          // 0..60
        async_ld128((unsigned)(size_t)&Bs[buf][k * LDB1 + nc],
                    Bw + ((size_t)kt * TK + k) * (size_t)ldB + n0 + nc);
      }
    }
  };

  const int nK = K / TK;
  issueTile(0, 0);

  for (int kt = 0; kt < nK; ++kt) {
    const int buf = kt & 1;
    wait_async0();          // own async loads for buf done
    __syncthreads();        // everyone's loads done; prev compute done
    if (kt + 1 < nK) issueTile(kt + 1, buf ^ 1);  // overlap DMA with math

    // ---- gather bf16 fragments (contiguous k = 16*half + s, same order for A and B) ----
    // Each aligned float4 chunk covers exactly two fragment dwords:
    // ds_load_b128 + 2x v_cvt_pk_bf16_f32, no byte-perms.
    BF16Frag afrag[2], bfrag[2];
#pragma unroll
    for (int sm = 0; sm < 2; ++sm) {
      const float* ar = &As[buf][(wm + sm * 16 + lr) * LDA_T + half * 16];
      const float4 f0 = *(const float4*)(ar);           // k = h16+0..3  -> u0,u1
      const float4 f1 = *(const float4*)(ar + 4);       // k = h16+4..7  -> u2,u3
      const float4 f2 = *(const float4*)(ar + 8);       // k = h16+8..11 -> u4,u5
      const float4 f3 = *(const float4*)(ar + 12);      // k = h16+12..15-> u6,u7
      afrag[sm].u[0] = packbf(f0.x, f0.y); afrag[sm].u[1] = packbf(f0.z, f0.w);
      afrag[sm].u[2] = packbf(f1.x, f1.y); afrag[sm].u[3] = packbf(f1.z, f1.w);
      afrag[sm].u[4] = packbf(f2.x, f2.y); afrag[sm].u[5] = packbf(f2.z, f2.w);
      afrag[sm].u[6] = packbf(f3.x, f3.y); afrag[sm].u[7] = packbf(f3.z, f3.w);
    }
    if (BLAYOUT == 0) {
#pragma unroll
      for (int sn = 0; sn < 2; ++sn) {
        const float* br = &Bs[buf][(wn + sn * 16 + lr) * LDB0 + half * 16];
        const float4 g0 = *(const float4*)(br);         // k = h16+0..3  -> u0,u1
        const float4 g1 = *(const float4*)(br + 4);     // k = h16+4..7  -> u2,u3
        const float4 g2 = *(const float4*)(br + 8);     // k = h16+8..11 -> u4,u5
        const float4 g3 = *(const float4*)(br + 12);    // k = h16+12..15-> u6,u7
        bfrag[sn].u[0] = packbf(g0.x, g0.y); bfrag[sn].u[1] = packbf(g0.z, g0.w);
        bfrag[sn].u[2] = packbf(g1.x, g1.y); bfrag[sn].u[3] = packbf(g1.z, g1.w);
        bfrag[sn].u[4] = packbf(g2.x, g2.y); bfrag[sn].u[5] = packbf(g2.z, g2.w);
        bfrag[sn].u[6] = packbf(g3.x, g3.y); bfrag[sn].u[7] = packbf(g3.z, g3.w);
      }
    } else {
#pragma unroll
      for (int sn = 0; sn < 2; ++sn) {
        int col = wn + sn * 16 + lr;
#pragma unroll
        for (int j = 0; j < 8; ++j) {
          int k = half * 16 + 2 * j;
          bfrag[sn].u[j] = packbf(Bs[buf][k * LDB1 + col],
                                  Bs[buf][(k + 1) * LDB1 + col]);
        }
      }
    }
#pragma unroll
    for (int sm = 0; sm < 2; ++sm)
#pragma unroll
      for (int sn = 0; sn < 2; ++sn)
        acc[sm][sn] = __builtin_amdgcn_wmma_f32_16x16x32_bf16(
            false, afrag[sm].v, false, bfrag[sn].v,
            (short)0, acc[sm][sn], false, false);
  }

  // ---- epilogue: 32-bit C/D layout (lane n = lr, VGPR r -> m = r + half*8) ----
#pragma unroll
  for (int sm = 0; sm < 2; ++sm) {
#pragma unroll
    for (int sn = 0; sn < 2; ++sn) {
      int nIdx = n0 + wn + sn * 16 + lr;
#pragma unroll
      for (int r = 0; r < 8; ++r) {
        int mIdx = m0 + wm + sm * 16 + half * 8 + r;
        float vacc = acc[sm][sn][r];
        if (EPI == 1)      vacc = vacc + E[(size_t)mIdx * ldE + nIdx];
        else if (EPI == 2) vacc = vacc * scale + E[(size_t)mIdx * ldE + nIdx];
        C[(size_t)mIdx * ldC + nIdx] = vacc;
      }
    }
  }
}

// ---------------- RMSNorm over D=2048 ----------------
__global__ __launch_bounds__(256)
void rmsnorm_kernel(const float* __restrict__ in, const float* __restrict__ w,
                    float* __restrict__ out, int D_,
                    long long inStrideB, long long outStrideB)
{
  const int b = blockIdx.y, row = blockIdx.x;
  const float* x = in + (size_t)b * inStrideB + (size_t)row * D_;
  float* y = out + (size_t)b * outStrideB + (size_t)row * D_;
  __shared__ float red[256];
  float ss = 0.f;
  for (int d = threadIdx.x; d < D_; d += 256) { float v = x[d]; ss += v * v; }
  red[threadIdx.x] = ss; __syncthreads();
  for (int s = 128; s > 0; s >>= 1) {
    if (threadIdx.x < s) red[threadIdx.x] += red[threadIdx.x + s];
    __syncthreads();
  }
  const float inv = rsqrtf(red[0] / (float)D_ + EPSV);
  for (int d = threadIdx.x; d < D_; d += 256) y[d] = x[d] * inv * w[d];
}

// ---------------- per-head RMSNorm + RoPE + [B,S,nh,DH]->[B,nh,S,DH] ----------------
__global__ __launch_bounds__(128)
void head_rope_kernel(const float* __restrict__ in,
                      const float* __restrict__ w0, const float* __restrict__ w1,
                      const float* __restrict__ w2,
                      const int* __restrict__ pos,
                      float* __restrict__ out, int nh, int doNorm)
{
  const int s = blockIdx.x, h = blockIdx.y, b = blockIdx.z, d = threadIdx.x;
  const float* x = in + (((size_t)b * SS + s) * nh + h) * DHH;
  float v = x[d];
  __shared__ float sh[DHH];
  __shared__ float red[DHH];
  if (doNorm) {
    red[d] = v * v; __syncthreads();
    for (int st = 64; st > 0; st >>= 1) {
      if (d < st) red[d] += red[d + st];
      __syncthreads();
    }
    const float* w = (s < SEQ0) ? w0 : ((s < SEQ0 + SEQ1) ? w1 : w2);
    v = v * rsqrtf(red[0] / (float)DHH + EPSV) * w[d];
    sh[d] = v; __syncthreads();
    const int p = pos[(size_t)b * SS + s];
    const int j = d & 63;
    const float ang = (float)p * expf(-(float)j * (LOGTHETA / 64.f));
    const float c = cosf(ang), sn = sinf(ang);
    const float rot = (d < 64) ? -sh[d + 64] : sh[d - 64];
    v = v * c + rot * sn;
  }
  out[(((size_t)b * nh + h) * SS + s) * DHH + d] = v;
}

// ---------------- row softmax over S=1344 (in place, fp32) ----------------
__global__ __launch_bounds__(256)
void softmax_kernel(float* __restrict__ sc)
{
  float* row = sc + ((size_t)blockIdx.y * SS + blockIdx.x) * SS;
  __shared__ float red[256];
  float mx = -1e30f;
  for (int t = threadIdx.x; t < SS; t += 256) mx = fmaxf(mx, row[t]);
  red[threadIdx.x] = mx; __syncthreads();
  for (int s = 128; s > 0; s >>= 1) {
    if (threadIdx.x < s) red[threadIdx.x] = fmaxf(red[threadIdx.x], red[threadIdx.x + s]);
    __syncthreads();
  }
  mx = red[0]; __syncthreads();
  float sum = 0.f;
  for (int t = threadIdx.x; t < SS; t += 256) {
    float e = __expf(row[t] - mx); row[t] = e; sum += e;
  }
  red[threadIdx.x] = sum; __syncthreads();
  for (int s = 128; s > 0; s >>= 1) {
    if (threadIdx.x < s) red[threadIdx.x] += red[threadIdx.x + s];
    __syncthreads();
  }
  const float inv = 1.f / red[0];
  for (int t = threadIdx.x; t < SS; t += 256) row[t] *= inv;
}

// ---------------- [B,H,S,DH] -> [B,S,H*DH] ----------------
__global__ __launch_bounds__(256)
void merge_heads_kernel(const float* __restrict__ ot, float* __restrict__ o)
{
  const size_t total = (size_t)BB * HH * SS * DHH;
  size_t i = (size_t)blockIdx.x * 256 + threadIdx.x;
  if (i >= total) return;
  int d = (int)(i & (DHH - 1));
  size_t r = i >> 7;
  int s = (int)(r % SS); r /= SS;
  int h = (int)(r % HH);
  int b = (int)(r / HH);
  o[(((size_t)b * SS + s) * HH + h) * DHH + d] = ot[i];
}

// ---------------- SwiGLU: g = silu(g) * u ----------------
__global__ __launch_bounds__(256)
void swiglu_kernel(float* __restrict__ g, const float* __restrict__ u, long long n)
{
  long long i = (long long)blockIdx.x * 256 + threadIdx.x;
  if (i >= n) return;
  float x = g[i];
  g[i] = x * (1.f / (1.f + __expf(-x))) * u[i];
}

// ---------------- host launcher ----------------
extern "C" void kernel_launch(void* const* d_in, const int* in_sizes, int n_in,
                              void* d_out, int out_size, void* d_ws, size_t ws_size,
                              hipStream_t stream)
{
  (void)in_sizes; (void)n_in; (void)out_size; (void)ws_size;
  const int SEQ[3]   = {SEQ0, SEQ1, SEQ2};
  const int START[3] = {0, SEQ0, SEQ0 + SEQ1};
  const int FF[3]    = {6144, 4096, 4096};

  const float *x[3], *ln1[3], *qw[3], *kw[3], *vw[3], *qn[3], *kn[3],
              *ow[3], *ln2[3], *gw[3], *uw[3], *dw[3];
  for (int i = 0; i < 3; ++i) {
    const int base = i * 12;
    x[i]   = (const float*)d_in[base + 0];
    ln1[i] = (const float*)d_in[base + 1];
    qw[i]  = (const float*)d_in[base + 2];
    kw[i]  = (const float*)d_in[base + 3];
    vw[i]  = (const float*)d_in[base + 4];
    qn[i]  = (const float*)d_in[base + 5];
    kn[i]  = (const float*)d_in[base + 6];
    ow[i]  = (const float*)d_in[base + 7];
    ln2[i] = (const float*)d_in[base + 8];
    gw[i]  = (const float*)d_in[base + 9];
    uw[i]  = (const float*)d_in[base + 10];
    dw[i]  = (const float*)d_in[base + 11];
  }
  const float* mask = (const float*)d_in[36];
  const int*   pos  = (const int*)d_in[37];
  float* out = (float*)d_out;
  float* ws  = (float*)d_ws;

  // workspace layout (fp32 elements)
  const long long SZ_H  = 2LL * 1024 * DD;
  const long long SZ_Q  = (long long)BB * SS * (HH * DHH);
  const long long SZ_KV = (long long)BB * SS * (KHH * DHH);
  const long long SZ_G  = 2LL * 1024 * 6144;
  float* hbuf = ws;
  float* qbuf = hbuf + SZ_H;
  float* kbuf = qbuf + SZ_Q;
  float* vbuf = kbuf + SZ_KV;
  float* qt   = vbuf + SZ_KV;
  float* kt   = qt   + SZ_Q;
  float* vt   = kt   + SZ_KV;
  float* ot   = vt   + SZ_KV;
  float* obuf = ot   + SZ_Q;
  float* r1   = obuf + SZ_Q;
  float* gbuf = r1   + SZ_Q;
  float* ubuf = gbuf + SZ_G;
  float* scores = ubuf + SZ_G;                      // B*H*S*S fp32

  // ===== Phase 1: per-expert RMSNorm + QKV projections =====
  for (int i = 0; i < 3; ++i) {
    const int s = SEQ[i], st = START[i];
    rmsnorm_kernel<<<dim3(s, BB), 256, 0, stream>>>(
        x[i], ln1[i], hbuf, DD, (long long)s * DD, (long long)s * DD);
    gemm_kernel<0, 0><<<dim3((HH * DHH) / 64, s / 64, BB), 128, 0, stream>>>(
        hbuf, qw[i], qbuf + (size_t)st * (HH * DHH), nullptr,
        DD, DD, DD, HH * DHH, 0,
        (long long)s * DD, 0, 0, 1, 1,
        (long long)SS * (HH * DHH), 0, 1, 1.f);
    gemm_kernel<0, 0><<<dim3((KHH * DHH) / 64, s / 64, BB), 128, 0, stream>>>(
        hbuf, kw[i], kbuf + (size_t)st * (KHH * DHH), nullptr,
        DD, DD, DD, KHH * DHH, 0,
        (long long)s * DD, 0, 0, 1, 1,
        (long long)SS * (KHH * DHH), 0, 1, 1.f);
    gemm_kernel<0, 0><<<dim3((KHH * DHH) / 64, s / 64, BB), 128, 0, stream>>>(
        hbuf, vw[i], vbuf + (size_t)st * (KHH * DHH), nullptr,
        DD, DD, DD, KHH * DHH, 0,
        (long long)s * DD, 0, 0, 1, 1,
        (long long)SS * (KHH * DHH), 0, 1, 1.f);
  }

  // ===== Phase 2: QK norm + RoPE + transpose; attention =====
  head_rope_kernel<<<dim3(SS, HH, BB), 128, 0, stream>>>(
      qbuf, qn[0], qn[1], qn[2], pos, qt, HH, 1);
  head_rope_kernel<<<dim3(SS, KHH, BB), 128, 0, stream>>>(
      kbuf, kn[0], kn[1], kn[2], pos, kt, KHH, 1);
  head_rope_kernel<<<dim3(SS, KHH, BB), 128, 0, stream>>>(
      vbuf, nullptr, nullptr, nullptr, pos, vt, KHH, 0);

  // scores[b,h] = SCALE * qt[b,h] @ kt[b,h/2]^T + mask[b]
  gemm_kernel<2, 0><<<dim3(SS / 64, SS / 64, BB * HH), 128, 0, stream>>>(
      qt, kt, scores, mask,
      DHH, DHH, DHH, SS, SS,
      (long long)SS * DHH, (long long)KHH * SS * DHH, (long long)SS * DHH,
      HH, HH / KHH,
      (long long)SS * SS, (long long)SS * SS, HH, SCALEV);

  softmax_kernel<<<dim3(SS, BB * HH), 256, 0, stream>>>(scores);

  // ot[b,h] = att[b,h] @ vt[b,h/2]  (B operand is [K=S, N=DH] layout)
  gemm_kernel<0, 1><<<dim3(DHH / 64, SS / 64, BB * HH), 128, 0, stream>>>(
      scores, vt, ot, nullptr,
      SS, SS, DHH, DHH, 0,
      (long long)SS * SS, (long long)KHH * SS * DHH, (long long)SS * DHH,
      HH, HH / KHH,
      (long long)SS * DHH, 0, 1, 1.f);

  {
    const long long tot = (long long)BB * HH * SS * DHH;
    merge_heads_kernel<<<(unsigned)((tot + 255) / 256), 256, 0, stream>>>(ot, obuf);
  }

  // ===== Phase 3: per-expert o_proj + MLP =====
  for (int i = 0; i < 3; ++i) {
    const int s = SEQ[i], st = START[i], ff = FF[i];
    gemm_kernel<1, 0><<<dim3(DD / 64, s / 64, BB), 128, 0, stream>>>(
        obuf + (size_t)st * (HH * DHH), ow[i], r1 + (size_t)st * DD, x[i],
        HH * DHH, HH * DHH, HH * DHH, DD, DD,
        (long long)SS * (HH * DHH), 0, 0, 1, 1,
        (long long)SS * DD, (long long)s * DD, 1, 1.f);
    rmsnorm_kernel<<<dim3(s, BB), 256, 0, stream>>>(
        r1 + (size_t)st * DD, ln2[i], hbuf, DD, (long long)SS * DD, (long long)s * DD);
    gemm_kernel<0, 0><<<dim3(ff / 64, s / 64, BB), 128, 0, stream>>>(
        hbuf, gw[i], gbuf, nullptr,
        DD, DD, DD, ff, 0,
        (long long)s * DD, 0, 0, 1, 1,
        (long long)s * ff, 0, 1, 1.f);
    gemm_kernel<0, 0><<<dim3(ff / 64, s / 64, BB), 128, 0, stream>>>(
        hbuf, uw[i], ubuf, nullptr,
        DD, DD, DD, ff, 0,
        (long long)s * DD, 0, 0, 1, 1,
        (long long)s * ff, 0, 1, 1.f);
    {
      const long long n = (long long)BB * s * ff;
      swiglu_kernel<<<(unsigned)((n + 255) / 256), 256, 0, stream>>>(gbuf, ubuf, n);
    }
    gemm_kernel<1, 0><<<dim3(DD / 64, s / 64, BB), 128, 0, stream>>>(
        gbuf, dw[i], out + (size_t)st * DD, r1 + (size_t)st * DD,
        ff, ff, ff, DD, DD,
        (long long)s * ff, 0, 0, 1, 1,
        (long long)SS * DD, (long long)SS * DD, 1, 1.f);
  }
}
